// HeteroGAT_9345848836471
// MI455X (gfx1250) — matrix operand order repaired and checked by reading, not compiled, not run on hardware
//
#include <hip/hip_runtime.h>
#include <hip/hip_bf16.h>
#include <math.h>

#define DH  96
#define DIN 32
#define GH  2
#define GD  48
#define HH  6
#define HD  16
#define NEG 0.2f
#define NT  6      // DH/16 column tiles

typedef __attribute__((ext_vector_type(16))) _Float16 v16h;
typedef __attribute__((ext_vector_type(8)))  float    v8f;

// ---------------------------------------------------------------- utilities
__global__ void fill_kernel(float* __restrict__ p, float v, int n) {
  int i = blockIdx.x * blockDim.x + threadIdx.x;
  if (i < n) p[i] = v;
}

__device__ inline void atomic_max_f32(float* addr, float val) {
  int* ia = (int*)addr;
  int cur = __float_as_int(*addr);
  while (__int_as_float(cur) < val) {
    int prev = atomicCAS(ia, cur, __float_as_int(val));
    if (prev == cur) break;
    cur = prev;
  }
}

// ---------------------------------------------------------------- WMMA GEMM
// C[M x 96] = A[M x K] * B[K x 96] (+bias). fp32 in/out, f16 fragments.
// Requires: M % 16 == 0, K % 32 == 0 (true for all call sites: K = 32 or 96),
// lda == K, ldb == ldc == 96.
// Block = 256 threads (8 waves). B is staged transposed in LDS as f16 so each
// B-fragment is one contiguous 32-byte LDS read. Each wave computes one
// 16-row stripe x all 6 column tiles, reusing a single A fragment per k-step.
__global__ void gemm_wmma(const float* __restrict__ A,
                          const float* __restrict__ B,
                          const float* __restrict__ bias,
                          float* __restrict__ C,
                          int M, int K) {
  __shared__ _Float16 Bt[DH * DH];   // [col][k] (k contiguous), K <= 96
  int tid = threadIdx.x;
  for (int idx = tid; idx < K * DH; idx += 256) {
    int k = idx / DH, c = idx - k * DH;
    Bt[c * K + k] = (_Float16)B[idx];          // B row-major: B[k*96 + c]
  }
  __syncthreads();

  int wave = tid >> 5, lane = tid & 31;
  int tm = blockIdx.x * 8 + wave;
  if (tm >= (M >> 4)) return;                  // no barriers after this point

  int arow   = (tm << 4) + (lane & 15);
  int ka_off = (lane >> 4) << 4;               // 0 or 16
  int kb_off = (lane >> 4) << 4;               // 0 or 16
  int lcol   = lane & 15;

  v8f zero = {};
  v8f acc[NT];
#pragma unroll
  for (int t = 0; t < NT; ++t) acc[t] = zero;

  for (int k0 = 0; k0 < K; k0 += 32) {
    const float* ap = A + (size_t)arow * K + k0 + ka_off;
    float4 a0 = *(const float4*)(ap);
    float4 a1 = *(const float4*)(ap + 4);
    float4 a2 = *(const float4*)(ap + 8);
    float4 a3 = *(const float4*)(ap + 12);
    v16h af;
    af[0]  = (_Float16)a0.x; af[1]  = (_Float16)a0.y;
    af[2]  = (_Float16)a0.z; af[3]  = (_Float16)a0.w;
    af[4]  = (_Float16)a1.x; af[5]  = (_Float16)a1.y;
    af[6]  = (_Float16)a1.z; af[7]  = (_Float16)a1.w;
    af[8]  = (_Float16)a2.x; af[9]  = (_Float16)a2.y;
    af[10] = (_Float16)a2.z; af[11] = (_Float16)a2.w;
    af[12] = (_Float16)a3.x; af[13] = (_Float16)a3.y;
    af[14] = (_Float16)a3.z; af[15] = (_Float16)a3.w;
#pragma unroll
    for (int t = 0; t < NT; ++t) {
      v16h bf = *(const v16h*)(&Bt[(t * 16 + lcol) * K + k0 + kb_off]);
      acc[t] = __builtin_amdgcn_wmma_f32_16x16x32_f16(false, af, false, bf,
                                                      (short)0, acc[t], false, false);
    }
  }

  int rbase = (tm << 4) + ((lane >> 4) << 3);
#pragma unroll
  for (int t = 0; t < NT; ++t) {
    int bcol = t * 16 + lcol;
    float bb = bias ? bias[bcol] : 0.0f;
#pragma unroll
    for (int r = 0; r < 8; ++r)
      C[(size_t)(rbase + r) * DH + bcol] = acc[t][r] + bb;
  }
}

// HGT relation transform: OUT[n, h*16+e] = sum_d KV[n, h*16+d] * REL[h][d][e]
// K=16 zero-padded to a 16x16x32 f16 WMMA, branchlessly (upper-K lane group
// multiplies its loaded fragment by 0; no exec-masked loads).
__global__ void hgt_rel_wmma(const float* __restrict__ KV,
                             const float* __restrict__ REL,
                             float* __restrict__ OUT, int n) {
  __shared__ _Float16 Rt[HH * HD * HD];        // [head][col][k]
  int tid = threadIdx.x;
  for (int idx = tid; idx < HH * HD * HD; idx += 256) {
    int hd = idx >> 8;
    int rem = idx & 255;
    int k = rem >> 4, c = rem & 15;
    Rt[(hd * HD + c) * HD + k] = (_Float16)REL[(hd * HD + k) * HD + c];
  }
  __syncthreads();

  int wave = tid >> 5, lane = tid & 31;
  int gw = blockIdx.x * 8 + wave;
  int tiles = (n >> 4) * HH;
  if (gw >= tiles) return;

  int head = gw % HH, tm = gw / HH;
  int arow   = (tm << 4) + (lane & 15);
  int ka_off = 0;
  int lcol   = lane & 15;
  int kgrp   = lane >> 4;                      // 1 => K 16..31 (all padding)

  const float* ap = KV + (size_t)arow * DH + head * HD + ka_off;
  float4 a0 = *(const float4*)(ap);
  float4 a1 = *(const float4*)(ap + 4);
  v16h af;
  af[0] = (_Float16)a0.x; af[1] = (_Float16)a0.y;
  af[2] = (_Float16)a0.z; af[3] = (_Float16)a0.w;
  af[4] = (_Float16)a1.x; af[5] = (_Float16)a1.y;
  af[6] = (_Float16)a1.z; af[7] = (_Float16)a1.w;
#pragma unroll
  for (int i = 8; i < 16; ++i) af[i] = (_Float16)ap[i];

  v16h bf = *(const v16h*)(&Rt[(head * HD + lcol) * HD]);
  _Float16 mask = (kgrp == 0) ? (_Float16)1.0f : (_Float16)0.0f;
#pragma unroll
  for (int i = 0; i < 16; ++i) bf[i] *= mask;

  v8f acc = {};
  acc = __builtin_amdgcn_wmma_f32_16x16x32_f16(false, af, false, bf,
                                               (short)0, acc, false, false);
  int rbase = (tm << 4) + ((lane >> 4) << 3);
#pragma unroll
  for (int r = 0; r < 8; ++r)
    OUT[(size_t)(rbase + r) * DH + head * HD + lcol] = acc[r];
}

// ---------------------------------------------------------------- GAT kernels
__global__ void gat_att_kernel(const float* __restrict__ hfeat,
                               const float* __restrict__ av_src,
                               const float* __restrict__ av_dst,
                               float* __restrict__ asrc,
                               float* __restrict__ adst, int n) {
  int i = blockIdx.x * blockDim.x + threadIdx.x;
  if (i >= n * GH) return;
  int node = i >> 1, hd = i & 1;
  const float* row = hfeat + node * DH + hd * GD;
  const float* as = av_src + hd * GD;
  const float* ad = av_dst + hd * GD;
  float s = 0.f, d = 0.f;
#pragma unroll 8
  for (int j = 0; j < GD; ++j) {
    float v = row[j];
    s = fmaf(v, as[j], s);
    d = fmaf(v, ad[j], d);
  }
  asrc[i] = s;
  adst[i] = d;
}

__global__ void gat_edge_max(const int* __restrict__ src, const int* __restrict__ dst,
                             const float* __restrict__ asrc, const float* __restrict__ adst,
                             float* __restrict__ segmax, float* __restrict__ elog,
                             int E_, int n) {
  int i = blockIdx.x * blockDim.x + threadIdx.x;
  if (i >= (E_ + n) * GH) return;
  int e = i >> 1, hd = i & 1;
  int s = (e < E_) ? src[e] : (e - E_);
  int d = (e < E_) ? dst[e] : (e - E_);
  float l = asrc[s * GH + hd] + adst[d * GH + hd];
  l = (l > 0.f) ? l : NEG * l;
  elog[i] = l;
  atomic_max_f32(&segmax[d * GH + hd], l);
}

__global__ void gat_edge_exp(const int* __restrict__ dst,
                             const float* __restrict__ segmax,
                             float* __restrict__ elog,
                             float* __restrict__ segsum, int E_, int n) {
  int i = blockIdx.x * blockDim.x + threadIdx.x;
  if (i >= (E_ + n) * GH) return;
  int e = i >> 1, hd = i & 1;
  int d = (e < E_) ? dst[e] : (e - E_);
  float ex = __expf(elog[i] - segmax[d * GH + hd]);
  elog[i] = ex;
  atomicAdd(&segsum[d * GH + hd], ex);
}

__global__ void gat_edge_scatter(const int* __restrict__ src, const int* __restrict__ dst,
                                 const float* __restrict__ hfeat,
                                 const float* __restrict__ elog,
                                 const float* __restrict__ segsum,
                                 float* __restrict__ outacc, int E_, int n) {
  int gw = (blockIdx.x * blockDim.x + threadIdx.x) >> 5;
  int lane = threadIdx.x & 31;
  if (gw >= E_ + n) return;
  int s = (gw < E_) ? src[gw] : (gw - E_);
  int d = (gw < E_) ? dst[gw] : (gw - E_);
#pragma unroll
  for (int j = 0; j < 3; ++j) {
    int f = lane + 32 * j;          // coalesced
    int hd = f / GD;
    float w = elog[gw * GH + hd] / (segsum[d * GH + hd] + 1e-16f);
    atomicAdd(&outacc[d * DH + f], hfeat[s * DH + f] * w);
  }
}

__global__ void add_bias(float* __restrict__ p, const float* __restrict__ b, int n) {
  int i = blockIdx.x * blockDim.x + threadIdx.x;
  if (i >= n * DH) return;
  p[i] += b[i % DH];
}

// ---------------------------------------------------------------- BatchNorm
__global__ void colstats(const float* __restrict__ x, float* __restrict__ sum,
                         float* __restrict__ sq, int n) {
  __shared__ float ss[256], qs[256];
  int c = blockIdx.x, t = threadIdx.x;
  float s = 0.f, q = 0.f;
  for (int r = t; r < n; r += blockDim.x) {
    float v = x[r * DH + c];
    s += v;
    q = fmaf(v, v, q);
  }
  ss[t] = s; qs[t] = q;
  __syncthreads();
  for (int o = 128; o > 0; o >>= 1) {
    if (t < o) { ss[t] += ss[t + o]; qs[t] += qs[t + o]; }
    __syncthreads();
  }
  if (t == 0) { sum[c] = ss[0]; sq[c] = qs[0]; }
}

// out = lrelu( bn(xin) [+ extra] [+ resid] )
__global__ void bn_apply(const float* __restrict__ xin, const float* __restrict__ sum,
                         const float* __restrict__ sq, const float* __restrict__ gamma,
                         const float* __restrict__ beta, const float* __restrict__ resid,
                         const float* __restrict__ extra, float* __restrict__ outp, int n) {
  int i = blockIdx.x * blockDim.x + threadIdx.x;
  if (i >= n * DH) return;
  int c = i % DH;
  float inv_n = 1.0f / (float)n;
  float m = sum[c] * inv_n;
  float var = sq[c] * inv_n - m * m;
  float v = gamma[c] * (xin[i] - m) * rsqrtf(var + 1e-5f) + beta[c];
  if (extra) v += extra[i];
  if (resid) v += resid[i];
  outp[i] = (v > 0.f) ? v : NEG * v;
}

// ---------------------------------------------------------------- HGT kernels
__global__ void hgt_edge_score(const int* __restrict__ src, const int* __restrict__ dst,
                               const float* __restrict__ q, const float* __restrict__ kr,
                               const float* __restrict__ prel,
                               float* __restrict__ segmax, float* __restrict__ elog, int E_) {
  int i = blockIdx.x * blockDim.x + threadIdx.x;
  if (i >= E_ * HH) return;
  int e = i / HH, hd = i - e * HH;
  int s = src[e], d = dst[e];
  const float* qd = q + (size_t)d * DH + hd * HD;
  const float* ks = kr + (size_t)s * DH + hd * HD;
  float a = 0.f;
#pragma unroll
  for (int j = 0; j < HD; ++j) a = fmaf(qd[j], ks[j], a);
  a *= prel[hd] * 0.25f;  // 1/sqrt(HD)
  elog[i] = a;
  atomic_max_f32(&segmax[d * HH + hd], a);
}

__global__ void hgt_edge_exp(const int* __restrict__ dst,
                             const float* __restrict__ segmax,
                             float* __restrict__ elog,
                             float* __restrict__ segsum, int E_) {
  int i = blockIdx.x * blockDim.x + threadIdx.x;
  if (i >= E_ * HH) return;
  int e = i / HH, hd = i - e * HH;
  int d = dst[e];
  float ex = __expf(elog[i] - segmax[d * HH + hd]);
  elog[i] = ex;
  atomicAdd(&segsum[d * HH + hd], ex);
}

__global__ void hgt_edge_scatter(const int* __restrict__ src, const int* __restrict__ dst,
                                 const float* __restrict__ vr,
                                 const float* __restrict__ elog,
                                 const float* __restrict__ segsum,
                                 float* __restrict__ agg, int E_) {
  int gw = (blockIdx.x * blockDim.x + threadIdx.x) >> 5;
  int lane = threadIdx.x & 31;
  if (gw >= E_) return;
  int s = src[gw], d = dst[gw];
#pragma unroll
  for (int j = 0; j < 3; ++j) {
    int f = lane + 32 * j;
    int hd = f >> 4;
    float w = elog[gw * HH + hd] / (segsum[d * HH + hd] + 1e-16f);
    atomicAdd(&agg[d * DH + f], vr[s * DH + f] * w);
  }
}

__global__ void gelu_kernel(const float* __restrict__ xin, float* __restrict__ outp, int n) {
  int i = blockIdx.x * blockDim.x + threadIdx.x;
  if (i >= n) return;
  float v = xin[i];
  float inner = 0.7978845608028654f * (v + 0.044715f * v * v * v);
  outp[i] = 0.5f * v * (1.0f + tanhf(inner));
}

__global__ void hgt_blend(float* __restrict__ outp, const float* __restrict__ xh,
                          const float* __restrict__ skip, int n) {
  int i = blockIdx.x * blockDim.x + threadIdx.x;
  if (i >= n) return;
  float sk = 1.0f / (1.0f + __expf(-skip[0]));
  outp[i] = sk * outp[i] + (1.0f - sk) * xh[i];
}

__global__ void out_proj(const float* __restrict__ h, const float* __restrict__ Wo,
                         const float* __restrict__ bo, float* __restrict__ outp, int n) {
  int i = blockIdx.x * blockDim.x + threadIdx.x;
  if (i >= n) return;
  float s = bo[0];
#pragma unroll 8
  for (int j = 0; j < DH; ++j) s = fmaf(h[i * DH + j], Wo[j], s);
  outp[i] = s;
}

// ---------------------------------------------------------------- launch
extern "C" void kernel_launch(void* const* d_in, const int* in_sizes, int n_in,
                              void* d_out, int out_size, void* d_ws, size_t ws_size,
                              hipStream_t stream) {
  (void)n_in; (void)out_size; (void)ws_size;
  const int N_ = in_sizes[0] / DIN;
  const int E_ = in_sizes[1] / 2;
  const float* x = (const float*)d_in[0];
  const int* src_[3] = { (const int*)d_in[1], (const int*)d_in[2], (const int*)d_in[3] };
  const int* dst_[3] = { src_[0] + E_, src_[1] + E_, src_[2] + E_ };

  // params leaves start at 4, jax pytree (sorted dict keys) order:
  //  bn: 4..9 (g0,b0,g1,b1,g2,b2) | bn_final: 10,11 | bn_mid: 12,13
  //  gat[l][t]: 14 + (l*3+t)*4 : [W, a_dst, a_src, b]  (36 leaves, 14..49)
  //  hgt_final: 50.. (a_rel,ba,bk,bq,bv,m_rel,p_rel,skip,wa,wk,wq,wv)
  //  hgt_mid:   62..73 | out: 74 (W), 75 (b) | proj '1': 76,77  '2': 78,79
  const int P = 4;
  const int BN0 = P, BNF = P + 6, BNM = P + 8, GAT0 = P + 10;
  const int HF = P + 46, HM = P + 58, OUTW = P + 70, PRJ1 = P + 72, PRJ2 = P + 74;
  auto in_f = [&](int i) { return (const float*)d_in[i]; };

  float* ws = (float*)d_ws;
  float* h    = ws; ws += (size_t)N_ * DH;
  float* hn   = ws; ws += (size_t)N_ * DH;   // accumulator (GAT total / HGT agg)
  float* tmp  = ws; ws += (size_t)N_ * DH;   // per-type GEMM out / gelu tmp
  float* kb   = ws; ws += (size_t)N_ * DH;   // k proj / inject proj
  float* qb   = ws; ws += (size_t)N_ * DH;
  float* vb   = ws; ws += (size_t)N_ * DH;
  float* krb  = ws; ws += (size_t)N_ * DH;
  float* vrb  = ws; ws += (size_t)N_ * DH;
  float* hgo  = ws; ws += (size_t)N_ * DH;   // HGT output
  float* asrc = ws; ws += (size_t)N_ * HH;
  float* adst = ws; ws += (size_t)N_ * HH;
  float* smax = ws; ws += (size_t)N_ * HH;
  float* ssum = ws; ws += (size_t)N_ * HH;
  float* eexp = ws; ws += (size_t)(E_ + N_) * HH;
  float* csum = ws; ws += DH;
  float* csq  = ws; ws += DH;

  const int TB = 256;
  auto nblk = [&](long n) { return (int)((n + TB - 1) / TB); };
  auto fill = [&](float* p, float v, long n) {
    fill_kernel<<<nblk(n), TB, 0, stream>>>(p, v, (int)n);
  };
  auto gemm = [&](const float* A, const float* B, const float* bias,
                  float* C, int M, int K) {
    int rowtiles = M / 16;
    gemm_wmma<<<(rowtiles + 7) / 8, TB, 0, stream>>>(A, B, bias, C, M, K);
  };

  auto run_gat = [&](const float* xin, int K, int layer) {
    fill(hn, 0.f, (long)N_ * DH);
    for (int t = 0; t < 3; ++t) {
      int base = GAT0 + (layer * 3 + t) * 4;
      const float* W  = in_f(base + 0);
      const float* aD = in_f(base + 1);
      const float* aS = in_f(base + 2);
      const float* b  = in_f(base + 3);
      gemm(xin, W, nullptr, tmp, N_, K);
      gat_att_kernel<<<nblk((long)N_ * GH), TB, 0, stream>>>(tmp, aS, aD, asrc, adst, N_);
      fill(smax, -INFINITY, (long)N_ * GH);
      fill(ssum, 0.f, (long)N_ * GH);
      long tot = (long)(E_ + N_) * GH;
      gat_edge_max<<<nblk(tot), TB, 0, stream>>>(src_[t], dst_[t], asrc, adst, smax, eexp, E_, N_);
      gat_edge_exp<<<nblk(tot), TB, 0, stream>>>(dst_[t], smax, eexp, ssum, E_, N_);
      gat_edge_scatter<<<nblk((long)(E_ + N_) * 32), TB, 0, stream>>>(
          src_[t], dst_[t], tmp, eexp, ssum, hn, E_, N_);
      add_bias<<<nblk((long)N_ * DH), TB, 0, stream>>>(hn, b, N_);
    }
  };

  auto run_hgt = [&](const float* xh, int base) {
    const float* a_rel = in_f(base + 0);
    const float* ba    = in_f(base + 1);
    const float* bk    = in_f(base + 2);
    const float* bq    = in_f(base + 3);
    const float* bv    = in_f(base + 4);
    const float* m_rel = in_f(base + 5);
    const float* p_rel = in_f(base + 6);
    const float* skip  = in_f(base + 7);
    const float* wa    = in_f(base + 8);
    const float* wk    = in_f(base + 9);
    const float* wq    = in_f(base + 10);
    const float* wv    = in_f(base + 11);
    gemm(xh, wk, bk, kb, N_, DH);
    gemm(xh, wq, bq, qb, N_, DH);
    gemm(xh, wv, bv, vb, N_, DH);
    fill(hn, 0.f, (long)N_ * DH);
    for (int r = 0; r < 3; ++r) {
      int tiles = (N_ / 16) * HH;
      hgt_rel_wmma<<<(tiles + 7) / 8, TB, 0, stream>>>(kb, a_rel + (size_t)r * HH * HD * HD, krb, N_);
      hgt_rel_wmma<<<(tiles + 7) / 8, TB, 0, stream>>>(vb, m_rel + (size_t)r * HH * HD * HD, vrb, N_);
      fill(smax, -INFINITY, (long)N_ * HH);
      fill(ssum, 0.f, (long)N_ * HH);
      long tot = (long)E_ * HH;
      hgt_edge_score<<<nblk(tot), TB, 0, stream>>>(src_[r], dst_[r], qb, krb, p_rel + r * HH,
                                                   smax, eexp, E_);
      hgt_edge_exp<<<nblk(tot), TB, 0, stream>>>(dst_[r], smax, eexp, ssum, E_);
      hgt_edge_scatter<<<nblk((long)E_ * 32), TB, 0, stream>>>(src_[r], dst_[r], vrb, eexp, ssum, hn, E_);
    }
    gelu_kernel<<<nblk((long)N_ * DH), TB, 0, stream>>>(hn, tmp, N_ * DH);
    gemm(tmp, wa, ba, hgo, N_, DH);
    hgt_blend<<<nblk((long)N_ * DH), TB, 0, stream>>>(hgo, xh, skip, N_ * DH);
  };

  auto bn_stats_apply = [&](const float* xin, const float* g, const float* bt,
                            const float* resid, const float* extra, float* outp) {
    colstats<<<DH, TB, 0, stream>>>(xin, csum, csq, N_);
    bn_apply<<<nblk((long)N_ * DH), TB, 0, stream>>>(xin, csum, csq, g, bt, resid, extra, outp, N_);
  };

  // ---- layer 0: h = lrelu(bn(gat0(x)))
  run_gat(x, DIN, 0);
  bn_stats_apply(hn, in_f(BN0 + 0), in_f(BN0 + 1), nullptr, nullptr, h);

  // ---- layer 1: hn = bn(gat1(h)) + x@Wp1 + bp1 ; h = lrelu(h + hn)
  run_gat(h, DH, 1);
  gemm(x, in_f(PRJ1), in_f(PRJ1 + 1), kb, N_, DIN);
  bn_stats_apply(hn, in_f(BN0 + 2), in_f(BN0 + 3), h, kb, h);
  // ---- mid-global HGT: h = lrelu(h + bn(hgt_mid(h)))
  run_hgt(h, HM);
  bn_stats_apply(hgo, in_f(BNM + 0), in_f(BNM + 1), h, nullptr, h);

  // ---- layer 2
  run_gat(h, DH, 2);
  gemm(x, in_f(PRJ2), in_f(PRJ2 + 1), kb, N_, DIN);
  bn_stats_apply(hn, in_f(BN0 + 4), in_f(BN0 + 5), h, kb, h);

  // ---- final HGT
  run_hgt(h, HF);
  bn_stats_apply(hgo, in_f(BNF + 0), in_f(BNF + 1), h, nullptr, h);

  // ---- output projection
  out_proj<<<nblk(N_), TB, 0, stream>>>(h, in_f(OUTW), in_f(OUTW + 1), (float*)d_out, N_);
}
